// STBlock_11381663334856
// MI455X (gfx1250) — matrix operand order repaired and checked by reading, not compiled, not run on hardware
//
#include <hip/hip_runtime.h>
#include <math.h>

typedef __attribute__((ext_vector_type(16))) __bf16 v16bf;
typedef __attribute__((ext_vector_type(8)))  __bf16 v8bf;
typedef __attribute__((ext_vector_type(8)))  float  v8f;
typedef __attribute__((ext_vector_type(4)))  unsigned int v4u;
typedef __attribute__((ext_vector_type(4)))  int v4i;

#define N_NODES 1024
#define B_DIM   32
#define C_DIM   32
#define T_OUT   7
#define M_TOT   (B_DIM * C_DIM * T_OUT)   // 7168
#define OUT_ELEMS (32 * 32 * 1024 * 13)   // 13,631,488
#define MW      128                        // m-columns per block tile

union FragBF { v16bf v; v4u q[2]; };

// ---------------------------------------------------------------------------
// CDNA5 async global->LDS copy (16B per call), ASYNCcnt-tracked.
// Builtin signature (probe-confirmed): (v4i AS1*, v4i AS3*, Ii, Ii).
// ---------------------------------------------------------------------------
typedef __attribute__((address_space(1))) v4i* gv4i_p;
typedef __attribute__((address_space(3))) v4i* lv4i_p;

__device__ __forceinline__ void async_copy16(const __bf16* gsrc, __bf16* ldst) {
#if __has_builtin(__builtin_amdgcn_global_load_async_to_lds_b128)
    __builtin_amdgcn_global_load_async_to_lds_b128(
        (gv4i_p)(gsrc), (lv4i_p)(ldst), 0, 0);
#else
    *(v4u*)ldst = *(const v4u*)gsrc;     // sync fallback: b128 load + ds store
#endif
}

__device__ __forceinline__ void wait_async_le2() {
#if __has_builtin(__builtin_amdgcn_s_wait_asynccnt)
    __builtin_amdgcn_s_wait_asynccnt(2);
#else
    asm volatile("s_wait_asynccnt 0x2" ::: "memory");
#endif
}

__device__ __forceinline__ void wait_async_le0() {
#if __has_builtin(__builtin_amdgcn_s_wait_asynccnt)
    __builtin_amdgcn_s_wait_asynccnt(0);
#else
    asm volatile("s_wait_asynccnt 0x0" ::: "memory");
#endif
}

// Raw split workgroup barrier: no implicit vmem-drain (unlike __syncthreads),
// only a compiler reordering fence. Hardware ordering of the LDS tile data is
// provided by the explicit s_wait_asynccnt before the barrier.
__device__ __forceinline__ void wg_barrier() {
#if __has_builtin(__builtin_amdgcn_s_barrier_signal) && __has_builtin(__builtin_amdgcn_s_barrier_wait)
    asm volatile("" ::: "memory");
    __builtin_amdgcn_s_barrier_signal(-1);
    __builtin_amdgcn_s_barrier_wait(-1);
    asm volatile("" ::: "memory");
#else
    asm volatile("s_barrier_signal -1\n\ts_barrier_wait -1" ::: "memory");
#endif
}

// ---------------------------------------------------------------------------
__global__ void zero_kernel(float* __restrict__ out, int n) {
    int i = blockIdx.x * 256 + threadIdx.x;
    if (i < n) out[i] = 0.0f;
}

// ---------------------------------------------------------------------------
// graph (f32) -> bf16 adjacency, plus transposed copy for the second branch.
// ---------------------------------------------------------------------------
__global__ void adjconv_kernel(const float* __restrict__ g,
                               __bf16* __restrict__ a,
                               __bf16* __restrict__ at) {
    int i = blockIdx.x * 256 + threadIdx.x;     // 1024*1024 threads
    int v = i >> 10, w = i & 1023;
    float val = g[i];
    a[i] = (__bf16)val;
    at[w * 1024 + v] = (__bf16)val;
}

// ---------------------------------------------------------------------------
// Fused dilated-inception (both sets) + tanh*sigmoid gate.
// Writes h0 transposed for the WMMA B-operand: h0[m][n], m = b*224 + c*7 + j.
// ---------------------------------------------------------------------------
__global__ __launch_bounds__(256) void incep_kernel(
    const float* __restrict__ x,
    const float* __restrict__ w10, const float* __restrict__ b10,
    const float* __restrict__ w11, const float* __restrict__ b11,
    const float* __restrict__ w12, const float* __restrict__ b12,
    const float* __restrict__ w13, const float* __restrict__ b13,
    const float* __restrict__ w20, const float* __restrict__ b20,
    const float* __restrict__ w21, const float* __restrict__ b21,
    const float* __restrict__ w22, const float* __restrict__ b22,
    const float* __restrict__ w23, const float* __restrict__ b23,
    __bf16* __restrict__ h0)
{
    __shared__ float xs[32 * 8 * 13];           // x[b, cin, n0..n0+7, 0..12]
    const int b  = blockIdx.x >> 7;
    const int n0 = (blockIdx.x & 127) << 3;
    const int tid = threadIdx.x;

    for (int idx = tid; idx < 32 * 8 * 13; idx += 256) {
        int cin = idx / 104, r = idx - cin * 104;
        int nl = r / 13, t = r - nl * 13;
        xs[idx] = x[((b * 32 + cin) * 1024 + n0 + nl) * 13 + t];
    }
    __syncthreads();

    const int cout = tid & 31;                  // concat channel = ki*8 + oc
    const int nl   = tid >> 5;
    const int ki   = cout >> 3;
    const int oc   = cout & 7;
    const int k    = (ki == 0) ? 2 : (ki == 1) ? 3 : (ki == 2) ? 6 : 7;
    const float* wp1 = (ki == 0) ? w10 : (ki == 1) ? w11 : (ki == 2) ? w12 : w13;
    const float* wp2 = (ki == 0) ? w20 : (ki == 1) ? w21 : (ki == 2) ? w22 : w23;
    const float* bp1 = (ki == 0) ? b10 : (ki == 1) ? b11 : (ki == 2) ? b12 : b13;
    const float* bp2 = (ki == 0) ? b20 : (ki == 1) ? b21 : (ki == 2) ? b22 : b23;

    float a1[7], a2[7];
    #pragma unroll
    for (int j = 0; j < 7; ++j) { a1[j] = 0.f; a2[j] = 0.f; }

    for (int cin = 0; cin < 32; ++cin) {
        const float* xr  = &xs[cin * 104 + nl * 13];
        const float* wr1 = wp1 + (oc * 32 + cin) * k;   // (8,32,1,k)
        const float* wr2 = wp2 + (oc * 32 + cin) * k;
        for (int tau = 0; tau < k; ++tau) {
            float wa = wr1[tau], wb = wr2[tau];
            const float* xt = xr + (7 - k) + tau;       // input t = 7-k+j+tau
            #pragma unroll
            for (int j = 0; j < 7; ++j) {
                float xv = xt[j];
                a1[j] += wa * xv;
                a2[j] += wb * xv;
            }
        }
    }
    const float bv1 = bp1[oc], bv2 = bp2[oc];
    #pragma unroll
    for (int j = 0; j < 7; ++j) {
        float f = tanhf(a1[j] + bv1);
        float g = 1.0f / (1.0f + expf(-(a2[j] + bv2)));
        h0[(b * 224 + cout * 7 + j) * 1024 + n0 + nl] = (__bf16)(f * g);
    }
}

// ---------------------------------------------------------------------------
// Graph propagation step via WMMA bf16, LDS-staged B operand:
//   Y[v,m] = sum_w adjb[v,w] * hin[m][w]   ;   hout[m][v] = 0.875*h + 0.125*Y
// Block = 8 waves = 128 rows(v) x 128 cols(m). B tile (128m x 32k = 8KB) is
// async-DMA'd to LDS double-buffered (2 chunks/thread, ASYNCcnt); the next
// tile's fill and the next A fragment overlap the current 8 WMMAs.
// ---------------------------------------------------------------------------
__global__ __launch_bounds__(256) void prop_kernel(
    const __bf16* __restrict__ adjb,
    const __bf16* __restrict__ hin,
    __bf16* __restrict__ hout)
{
    __shared__ __align__(16) __bf16 bs[2][MW * 32];  // [buf][m_local*32 + k_local]

    const int tid  = threadIdx.x;
    const int lane = tid & 31;
    const int wave = tid >> 5;
    const int vblk = blockIdx.x & 7;    // 8 v-blocks of 128 rows
    const int mg   = blockIdx.x >> 3;   // 56 m-groups of 128 cols
    const int v0   = vblk * 128 + wave * 16;
    const int m0   = mg * MW;

    // async fill: each of 256 threads moves two 16B chunks of the 8KB tile
    const int c0 = tid, c1 = tid + 256;              // chunk ids (512 total)
    const int fm0 = c0 >> 2, fk0 = (c0 & 3) * 8;
    const int fm1 = c1 >> 2, fk1 = (c1 & 3) * 8;
    const __bf16* g0 = hin + (m0 + fm0) * 1024 + fk0;
    const __bf16* g1 = hin + (m0 + fm1) * 1024 + fk1;
    const int l0 = fm0 * 32 + fk0, l1 = fm1 * 32 + fk1;

    // A layout (matches B): lanes 0-15 -> K 0..15, lanes 16-31 -> K 16..31
    const int arow = v0 + (lane & 15);
    const int asel = (lane >> 4) ? 16 : 0;
    // B layout (matches A): lanes 0-15 -> K 0..15, lanes 16-31 -> K 16..31
    const int bsel = (lane >> 4) ? 16 : 0;
    const int msub = lane & 15;

    v8f acc[8];
    const v8f vz = {0.f, 0.f, 0.f, 0.f, 0.f, 0.f, 0.f, 0.f};
    #pragma unroll
    for (int t = 0; t < 8; ++t) acc[t] = vz;

    // prologue: DMA buffer 0 (k0 = 0), preload first A fragment
    async_copy16(g0, &bs[0][l0]);
    async_copy16(g1, &bs[0][l1]);
    FragBF Acur;
    {
        const __bf16* ap = adjb + arow * 1024 + asel;
        Acur.q[0] = *(const v4u*)(ap);
        Acur.q[1] = *(const v4u*)(ap + 8);
    }

    #pragma unroll 1
    for (int i = 0; i < 32; ++i) {
        const int k0 = i * 32;
        if (i + 1 < 32) {
            const int nb = (i + 1) & 1;
            async_copy16(g0 + (k0 + 32), &bs[nb][l0]);   // fill next tile
            async_copy16(g1 + (k0 + 32), &bs[nb][l1]);
            wait_async_le2();                            // current tile done
        } else {
            wait_async_le0();
        }
        wg_barrier();                    // tile visible to all 8 waves

        const __bf16* bb = &bs[i & 1][0];
        #pragma unroll
        for (int t = 0; t < 8; ++t) {
            v16bf Bv = *(const v16bf*)(bb + (msub + t * 16) * 32 + bsel);
            acc[t] = __builtin_amdgcn_wmma_f32_16x16x32_bf16(
                false, Acur.v, false, Bv, (short)0, acc[t], false, false);
        }
        if (i + 1 < 32) {                // reload A after its consumers (WAR ok)
            const __bf16* ap = adjb + arow * 1024 + (k0 + 32) + asel;
            Acur.q[0] = *(const v4u*)(ap);
            Acur.q[1] = *(const v4u*)(ap + 8);
        }
        wg_barrier();                    // reads done before next overwrite
    }

    // C layout: VGPR r holds row v0 + r (+8 for lanes 16-31), col = m.
    const int vbase = v0 + ((lane >> 4) ? 8 : 0);
    #pragma unroll
    for (int t = 0; t < 8; ++t) {
        const int m = m0 + msub + t * 16;
        v8bf hold = *(const v8bf*)(hin + m * 1024 + vbase);
        v8bf hnew;
        #pragma unroll
        for (int r = 0; r < 8; ++r) {
            float hv = 0.875f * (float)hold[r] + 0.125f * acc[t][r];
            hnew[r] = (__bf16)hv;
        }
        *(v8bf*)(hout + m * 1024 + vbase) = hnew;
    }
}

// ---------------------------------------------------------------------------
// Per-state channel MLP accumulation (channel MLP commutes with propagation):
//   out[b,o,v,6+j] += sum_c W[o, s*32+c] * state[m(b,c,j)][v]   (+ bias once)
// ---------------------------------------------------------------------------
__global__ __launch_bounds__(256) void mlpacc_kernel(
    const __bf16* __restrict__ state,
    const float* __restrict__ W,      // (32, 160)
    const float* __restrict__ bias,   // (32,)
    int s, int addb,
    float* __restrict__ out)
{
    __shared__ float ws[1024];
    const int tid = threadIdx.x;
    const int chunk = blockIdx.x & 3;
    const int j = (blockIdx.x >> 2) % 7;
    const int b = blockIdx.x / 28;

    for (int idx = tid; idx < 1024; idx += 256) {
        int o = idx >> 5, c = idx & 31;
        ws[idx] = W[o * 160 + s * 32 + c];
    }
    __syncthreads();

    const int v = chunk * 256 + tid;
    float st[32];
    #pragma unroll
    for (int c = 0; c < 32; ++c)
        st[c] = (float)state[(b * 224 + c * 7 + j) * 1024 + v];

    for (int o = 0; o < 32; ++o) {
        float acc = addb ? bias[o] : 0.0f;
        #pragma unroll
        for (int c = 0; c < 32; ++c) acc += ws[o * 32 + c] * st[c];
        size_t oi = ((size_t)((b * 32 + o) * 1024 + v)) * 13 + 6 + j;
        out[oi] += acc;
    }
}

// ---------------------------------------------------------------------------
extern "C" void kernel_launch(void* const* d_in, const int* in_sizes, int n_in,
                              void* d_out, int out_size, void* d_ws, size_t ws_size,
                              hipStream_t stream) {
    const float* x     = (const float*)d_in[0];
    const float* graph = (const float*)d_in[1];
    const float* i1w[4] = {(const float*)d_in[2], (const float*)d_in[4],
                           (const float*)d_in[6], (const float*)d_in[8]};
    const float* i1b[4] = {(const float*)d_in[3], (const float*)d_in[5],
                           (const float*)d_in[7], (const float*)d_in[9]};
    const float* i2w[4] = {(const float*)d_in[10], (const float*)d_in[12],
                           (const float*)d_in[14], (const float*)d_in[16]};
    const float* i2b[4] = {(const float*)d_in[11], (const float*)d_in[13],
                           (const float*)d_in[15], (const float*)d_in[17]};
    const float* mlp1w = (const float*)d_in[18];
    const float* mlp1b = (const float*)d_in[19];
    const float* mlp2w = (const float*)d_in[20];
    const float* mlp2b = (const float*)d_in[21];
    float* out = (float*)d_out;

    char* ws = (char*)d_ws;
    const size_t ADJ_BYTES   = (size_t)1024 * 1024 * 2;      // 2 MB
    const size_t STATE_BYTES = (size_t)M_TOT * 1024 * 2;     // 14.68 MB
    __bf16* adjb  = (__bf16*)(ws);
    __bf16* adjbT = (__bf16*)(ws + ADJ_BYTES);
    __bf16* h0    = (__bf16*)(ws + 2 * ADJ_BYTES);
    __bf16* p0    = (__bf16*)(ws + 2 * ADJ_BYTES + STATE_BYTES);
    __bf16* p1    = (__bf16*)(ws + 2 * ADJ_BYTES + 2 * STATE_BYTES);

    zero_kernel<<<(OUT_ELEMS + 255) / 256, 256, 0, stream>>>(out, OUT_ELEMS);
    adjconv_kernel<<<(1024 * 1024) / 256, 256, 0, stream>>>(graph, adjb, adjbT);
    incep_kernel<<<32 * 128, 256, 0, stream>>>(x,
        i1w[0], i1b[0], i1w[1], i1b[1], i1w[2], i1b[2], i1w[3], i1b[3],
        i2w[0], i2b[0], i2w[1], i2b[1], i2w[2], i2b[2], i2w[3], i2b[3], h0);

    const int PROP_BLOCKS = 8 * (M_TOT / MW);      // 8 v-blocks * 56 m-groups
    const int ACC_BLOCKS  = 32 * 7 * 4;            // 896

    // Branch A: adj, mlp1
    mlpacc_kernel<<<ACC_BLOCKS, 256, 0, stream>>>(h0, mlp1w, mlp1b, 0, 1, out);
    prop_kernel<<<PROP_BLOCKS, 256, 0, stream>>>(adjb, h0, p0);
    mlpacc_kernel<<<ACC_BLOCKS, 256, 0, stream>>>(p0, mlp1w, mlp1b, 1, 0, out);
    prop_kernel<<<PROP_BLOCKS, 256, 0, stream>>>(adjb, p0, p1);
    mlpacc_kernel<<<ACC_BLOCKS, 256, 0, stream>>>(p1, mlp1w, mlp1b, 2, 0, out);
    prop_kernel<<<PROP_BLOCKS, 256, 0, stream>>>(adjb, p1, p0);
    mlpacc_kernel<<<ACC_BLOCKS, 256, 0, stream>>>(p0, mlp1w, mlp1b, 3, 0, out);
    prop_kernel<<<PROP_BLOCKS, 256, 0, stream>>>(adjb, p0, p1);
    mlpacc_kernel<<<ACC_BLOCKS, 256, 0, stream>>>(p1, mlp1w, mlp1b, 4, 0, out);

    // Branch B: adj^T, mlp2 (h0 untouched above, reuse p0/p1)
    mlpacc_kernel<<<ACC_BLOCKS, 256, 0, stream>>>(h0, mlp2w, mlp2b, 0, 1, out);
    prop_kernel<<<PROP_BLOCKS, 256, 0, stream>>>(adjbT, h0, p0);
    mlpacc_kernel<<<ACC_BLOCKS, 256, 0, stream>>>(p0, mlp2w, mlp2b, 1, 0, out);
    prop_kernel<<<PROP_BLOCKS, 256, 0, stream>>>(adjbT, p0, p1);
    mlpacc_kernel<<<ACC_BLOCKS, 256, 0, stream>>>(p1, mlp2w, mlp2b, 2, 0, out);
    prop_kernel<<<PROP_BLOCKS, 256, 0, stream>>>(adjbT, p1, p0);
    mlpacc_kernel<<<ACC_BLOCKS, 256, 0, stream>>>(p0, mlp2w, mlp2b, 3, 0, out);
    prop_kernel<<<PROP_BLOCKS, 256, 0, stream>>>(adjbT, p0, p1);
    mlpacc_kernel<<<ACC_BLOCKS, 256, 0, stream>>>(p1, mlp2w, mlp2b, 4, 0, out);
}
